// TransformerDecoderLayer_MOE_66022237274328
// MI455X (gfx1250) — matrix operand order repaired and checked by reading, not compiled, not run on hardware
//
#include <hip/hip_runtime.h>
#include <hip/hip_bf16.h>

// ---------------- problem constants (from reference) ----------------
constexpr int S_ = 1024, B_ = 8, D_ = 1024, H_ = 16, HD_ = 64, F_ = 4096, E_ = 8;
constexpr int N_ = S_ * B_;            // 8192 tokens
constexpr int CAP_ = 4 * N_ / E_;      // 4096 capacity per expert

typedef __attribute__((ext_vector_type(16))) __bf16 v16bf;
typedef __attribute__((ext_vector_type(8)))  float  v8f;

union Frag {
    uint4 q[2];
    v16bf v;
};

// ---------------- small helpers ----------------
__device__ __forceinline__ unsigned short f2bf(float f) {
    unsigned u = __float_as_uint(f);
    unsigned r = u + 0x7FFFu + ((u >> 16) & 1u);
    return (unsigned short)(r >> 16);
}
__device__ __forceinline__ float bf2f(unsigned short h) {
    return __uint_as_float(((unsigned)h) << 16);
}

// A-fragment (16x32, 16-bit): lane L holds row M=L&15; elements i = K(2*kb+i),
// kb = (L<16)?0:8. rowp points at K-offset 0.
__device__ __forceinline__ Frag ldfragA(const unsigned short* rowp, int kb) {
    Frag f;
    f.q[0] = *(const uint4*)(rowp + 2 * kb);
    f.q[1] = *(const uint4*)(rowp + 2 * kb + 8);
    return f;
}
// B-fragment (32x16, 16-bit): lane L holds col n=L&15; elements i = K(kb2+i),
// kb2 = (L<16)?0:16. p points at (column n, K-offset kb2) of a K-contiguous layout.
__device__ __forceinline__ Frag ldfragB(const unsigned short* p) {
    Frag f;
    f.q[0] = *(const uint4*)p;
    f.q[1] = *(const uint4*)(p + 8);
    return f;
}
__device__ __forceinline__ v8f wmma_bf16(const Frag& a, const Frag& b, v8f c) {
    return __builtin_amdgcn_wmma_f32_16x16x32_bf16(false, a.v, false, b.v,
                                                   (short)0, c, false, false);
}

// ---------------- f32 -> bf16 conversion ----------------
__global__ void k_f2bf(const float* __restrict__ src, unsigned short* __restrict__ dst, size_t n) {
    size_t i = (size_t)blockIdx.x * blockDim.x + threadIdx.x;
    size_t stride = (size_t)gridDim.x * blockDim.x;
    for (; i < n; i += stride) dst[i] = f2bf(src[i]);
}

// ---------------- LayerNorm (one 256-thread block per row) ----------------
__global__ __launch_bounds__(256) void k_ln(const float* __restrict__ x,
                                            const float* __restrict__ g,
                                            const float* __restrict__ bta,
                                            unsigned short* __restrict__ out) {
    int r = blockIdx.x;
    const float* xr = x + (size_t)r * D_;
    float s = 0.f, s2 = 0.f;
    for (int i = threadIdx.x; i < D_; i += 256) { float v = xr[i]; s += v; s2 += v * v; }
    for (int d = 16; d >= 1; d >>= 1) { s += __shfl_xor(s, d, 32); s2 += __shfl_xor(s2, d, 32); }
    __shared__ float sh[16];
    int w = threadIdx.x >> 5, lane = threadIdx.x & 31;
    if (lane == 0) { sh[w] = s; sh[8 + w] = s2; }
    __syncthreads();
    if (threadIdx.x == 0) {
        float a = 0.f, bb = 0.f;
        for (int i = 0; i < 8; i++) { a += sh[i]; bb += sh[8 + i]; }
        sh[0] = a; sh[8] = bb;
    }
    __syncthreads();
    float mean = sh[0] / (float)D_;
    float var  = sh[8] / (float)D_ - mean * mean;
    float inv  = rsqrtf(var + 1e-5f);
    for (int i = threadIdx.x; i < D_; i += 256) {
        float v = (xr[i] - mean) * inv * g[i] + bta[i];
        out[(size_t)r * D_ + i] = f2bf(v);
    }
}

// ---------------- generic bf16 WMMA GEMM: C[128x128 tiles] = A(MxK) @ B(KxN) ----------
// MODE 0: outBF = bf16(acc*alpha + bias)                      (Q/K/V proj)
// MODE 1: outF  = resid + acc + bias                          (O proj + residual)
// MODE 2: outBF = bf16(relu(acc + bias))                      (expert FFN1)
// MODE 3: tok=slot2tok[row]; if tok>=0: outF[tok] += (acc+bias)*gateval[tok]  (FFN2+combine)
constexpr int BSTR = 40;   // padded LDS row stride (elements); 80B keeps 16B alignment

template <int MODE>
__global__ __launch_bounds__(256) void k_gemm(const unsigned short* __restrict__ A, int lda,
                                              const unsigned short* __restrict__ Bm, int ldb,
                                              const float* __restrict__ bias, float alpha,
                                              const float* __restrict__ resid,
                                              float* __restrict__ outF,
                                              unsigned short* __restrict__ outBF, int ldc,
                                              const int* __restrict__ slot2tok,
                                              const float* __restrict__ gateval, int K) {
    __shared__ unsigned short BT[128 * BSTR];   // B tile, transposed: [n][kk]
    int tid = threadIdx.x;
    int w = tid >> 5, lane = tid & 31;
    int n0 = blockIdx.x * 128, m0 = blockIdx.y * 128;
    int hi  = (lane < 16) ? 0 : 1;
    int kb  = hi ? 8 : 0;
    int kb2 = hi ? 16 : 0;
    int mb  = hi ? 8 : 0;
    int nl  = lane & 15;
    int mrow = m0 + w * 16 + nl;
    v8f vz = {0.f, 0.f, 0.f, 0.f, 0.f, 0.f, 0.f, 0.f};
    v8f acc[8];
#pragma unroll
    for (int t = 0; t < 8; t++) acc[t] = vz;
    const unsigned short* Arow = A + (size_t)mrow * lda;

    for (int k0 = 0; k0 < K; k0 += 32) {
        __syncthreads();
#pragma unroll
        for (int j = 0; j < 16; j++) {
            int idx = tid + j * 256;
            int kk = idx >> 7, n = idx & 127;
            BT[n * BSTR + kk] = Bm[(size_t)(k0 + kk) * ldb + n0 + n];
        }
        __syncthreads();
        if (k0 + 32 < K)
            __builtin_prefetch(&Bm[(size_t)(k0 + 32) * ldb + n0 + (tid & 127)], 0, 1);
        Frag a = ldfragA(Arow + k0, kb);
#pragma unroll
        for (int t = 0; t < 8; t++) {
            Frag b = ldfragB(&BT[(t * 16 + nl) * BSTR + kb2]);
            acc[t] = wmma_bf16(a, b, acc[t]);
        }
    }

    int tok[8];
    if constexpr (MODE == 3) {
#pragma unroll
        for (int j = 0; j < 8; j++) tok[j] = slot2tok[m0 + w * 16 + mb + j];
    }
#pragma unroll
    for (int t = 0; t < 8; t++) {
        int n = n0 + t * 16 + nl;
        float bn = bias[n];
#pragma unroll
        for (int j = 0; j < 8; j++) {
            int m = m0 + w * 16 + mb + j;
            float vv = acc[t][j];
            if constexpr (MODE == 0) {
                outBF[(size_t)m * ldc + n] = f2bf(vv * alpha + bn);
            } else if constexpr (MODE == 1) {
                outF[(size_t)m * ldc + n] = resid[(size_t)m * ldc + n] + vv + bn;
            } else if constexpr (MODE == 2) {
                float o = vv + bn;
                outBF[(size_t)m * ldc + n] = f2bf(o > 0.f ? o : 0.f);
            } else {
                int tk = tok[j];
                if (tk >= 0) {
                    size_t oi = (size_t)tk * ldc + n;
                    outF[oi] += (vv + bn) * gateval[tk];
                }
            }
        }
    }
}

// ---------------- flash attention: one wave per (16 q rows, b, h) ----------------
constexpr int KSs = 72, VSs = 40, PSs = 40;   // padded LDS strides (16B-aligned rows)

__global__ __launch_bounds__(32) void k_attn(const unsigned short* __restrict__ q,
                                             const unsigned short* __restrict__ k,
                                             const unsigned short* __restrict__ v,
                                             unsigned short* __restrict__ o) {
    __shared__ unsigned short LK[32 * KSs];    // k block: [key][hd]
    __shared__ unsigned short LVT[64 * VSs];   // v block transposed: [hd][key]
    __shared__ unsigned short LP[16 * PSs];    // prob tile: [qrow][key]
    int lane = threadIdx.x;
    int qb = blockIdx.x * 16;
    int bh = blockIdx.y;
    int b = bh >> 4, h = bh & 15;              // H_ == 16
    int hi  = (lane < 16) ? 0 : 1;
    int kb  = hi ? 8 : 0;
    int kb2 = hi ? 16 : 0;
    int mb  = hi ? 8 : 0;
    int nn  = lane & 15;
    auto idx = [&](int s, int hd) { return (size_t)((s * B_ + b) * D_ + h * HD_ + hd); };

    const unsigned short* qrow = q + idx(qb + nn, 0);
    Frag aq0 = ldfragA(qrow, kb);
    Frag aq1 = ldfragA(qrow + 32, kb);

    v8f vz = {0.f, 0.f, 0.f, 0.f, 0.f, 0.f, 0.f, 0.f};
    v8f oacc[4];
    float rmax[8], rsum[8];
#pragma unroll
    for (int j = 0; j < 8; j++) { rmax[j] = -1e30f; rsum[j] = 0.f; }
#pragma unroll
    for (int tt = 0; tt < 4; tt++) oacc[tt] = vz;

    int kend = qb + 16;                         // causal: keys < qb+16 needed
    for (int k0 = 0; k0 < kend; k0 += 32) {
        // cooperative load: lane owns one key row (k0+lane < S always; keys past qb+15 are masked)
        int key = k0 + lane;
        const unsigned short* kp = k + idx(key, 0);
        const unsigned short* vp = v + idx(key, 0);
        uint4 vr[8];
#pragma unroll
        for (int j = 0; j < 8; j++) {
            uint4 kq = *(const uint4*)(kp + 8 * j);
            *(uint4*)&LK[lane * KSs + 8 * j] = kq;
            vr[j] = *(const uint4*)(vp + 8 * j);
        }
        const unsigned short* vs = (const unsigned short*)vr;
#pragma unroll
        for (int hd = 0; hd < 64; hd++) LVT[hd * VSs + lane] = vs[hd];

        // scores: 16 q rows x 32 keys, K-dim = 64 (two WMMAs per 16-key subtile)
        v8f sc[2];
#pragma unroll
        for (int t = 0; t < 2; t++) {
            Frag b0 = ldfragB(&LK[(t * 16 + nn) * KSs + kb2]);
            Frag b1 = ldfragB(&LK[(t * 16 + nn) * KSs + 32 + kb2]);
            v8f s = wmma_bf16(aq0, b0, vz);
            sc[t] = wmma_bf16(aq1, b1, s);
        }
        // causal mask + online softmax (stats per q-row, reduced across 16 lanes)
#pragma unroll
        for (int j = 0; j < 8; j++) {
            int qr = qb + mb + j;
            float s0 = sc[0][j]; if (k0 + nn > qr)      s0 = -1e30f;
            float s1 = sc[1][j]; if (k0 + 16 + nn > qr) s1 = -1e30f;
            float mm = fmaxf(s0, s1);
            mm = fmaxf(mm, __shfl_xor(mm, 1, 32));
            mm = fmaxf(mm, __shfl_xor(mm, 2, 32));
            mm = fmaxf(mm, __shfl_xor(mm, 4, 32));
            mm = fmaxf(mm, __shfl_xor(mm, 8, 32));
            float nm = fmaxf(rmax[j], mm);
            float scl = __expf(rmax[j] - nm);
            float p0 = __expf(s0 - nm), p1 = __expf(s1 - nm);
            LP[(mb + j) * PSs + nn]      = f2bf(p0);
            LP[(mb + j) * PSs + 16 + nn] = f2bf(p1);
            float ps = p0 + p1;
            ps += __shfl_xor(ps, 1, 32);
            ps += __shfl_xor(ps, 2, 32);
            ps += __shfl_xor(ps, 4, 32);
            ps += __shfl_xor(ps, 8, 32);
            rsum[j] = rsum[j] * scl + ps;
            rmax[j] = nm;
#pragma unroll
            for (int tt = 0; tt < 4; tt++) oacc[tt][j] *= scl;
        }
        // O += P(16x32) @ V(32x64) : P re-read from LDS in A-layout
        Frag ap = ldfragA(&LP[nn * PSs], kb);
#pragma unroll
        for (int tt = 0; tt < 4; tt++) {
            Frag bv = ldfragB(&LVT[(tt * 16 + nn) * VSs + kb2]);
            oacc[tt] = wmma_bf16(ap, bv, oacc[tt]);
        }
    }
#pragma unroll
    for (int tt = 0; tt < 4; tt++)
#pragma unroll
        for (int j = 0; j < 8; j++) {
            float val = oacc[tt][j] / rsum[j];
            o[idx(qb + mb + j, tt * 16 + nn)] = f2bf(val);
        }
}

// ---------------- gating: logits = xn @ gate_w (N x 8) ----------------
__global__ __launch_bounds__(256) void k_gate_logits(const unsigned short* __restrict__ xn,
                                                     const unsigned short* __restrict__ gw,
                                                     float* __restrict__ logits) {
    int t = blockIdx.x * 256 + threadIdx.x;
    float acc[E_];
#pragma unroll
    for (int e = 0; e < E_; e++) acc[e] = 0.f;
    const unsigned short* xr = xn + (size_t)t * D_;
    for (int i = 0; i < D_; i++) {
        float xv = bf2f(xr[i]);
        uint4 gq = *(const uint4*)(gw + (size_t)i * E_);
        const unsigned short* gs = (const unsigned short*)&gq;
#pragma unroll
        for (int e = 0; e < E_; e++) acc[e] += xv * bf2f(gs[e]);
    }
#pragma unroll
    for (int e = 0; e < E_; e++) logits[(size_t)t * E_ + e] = acc[e];
}

__global__ void k_init(int* __restrict__ slot2tok, float* __restrict__ sums) {
    int i = blockIdx.x * blockDim.x + threadIdx.x;
    if (i < E_ * CAP_) slot2tok[i] = -1;
    if (i < 16) sums[i] = 0.f;
}

__global__ __launch_bounds__(256) void k_gate_soft(const float* __restrict__ logits,
                                                   int* __restrict__ eidx,
                                                   float* __restrict__ prob,
                                                   float* __restrict__ sums) {
    int t = blockIdx.x * 256 + threadIdx.x;
    float l[E_];
    float mx = -1e30f; int am = 0;
#pragma unroll
    for (int e = 0; e < E_; e++) {
        l[e] = logits[(size_t)t * E_ + e];
        if (l[e] > mx) { mx = l[e]; am = e; }
    }
    float s = 0.f;
#pragma unroll
    for (int e = 0; e < E_; e++) { l[e] = __expf(l[e] - mx); s += l[e]; }
    float inv = 1.f / s;
#pragma unroll
    for (int e = 0; e < E_; e++) atomicAdd(&sums[e], l[e] * inv);
    atomicAdd(&sums[8 + am], 1.0f);
    eidx[t] = am;
    prob[t] = l[am] * inv;
}

__global__ __launch_bounds__(32) void k_count(const int* __restrict__ eidx,
                                              int* __restrict__ waveCounts) {
    int c = blockIdx.x, lane = threadIdx.x;
    int e = eidx[c * 32 + lane];
#pragma unroll
    for (int ee = 0; ee < E_; ee++) {
        unsigned long long bm = __ballot(e == ee);
        if (lane == 0) waveCounts[c * E_ + ee] = __popcll(bm);
    }
}

__global__ void k_scan(const int* __restrict__ waveCounts, int* __restrict__ chunkBase) {
    int e = threadIdx.x;
    if (e >= E_) return;
    int run = 0;
    for (int c = 0; c < N_ / 32; c++) {
        chunkBase[c * E_ + e] = run;
        run += waveCounts[c * E_ + e];
    }
}

__global__ __launch_bounds__(32) void k_meta(const int* __restrict__ eidx,
                                             const float* __restrict__ prob,
                                             const int* __restrict__ chunkBase,
                                             float* __restrict__ gate_val,
                                             int* __restrict__ slot_keep,
                                             int* __restrict__ slot2tok) {
    int c = blockIdx.x, lane = threadIdx.x;
    int t = c * 32 + lane;
    int e = eidx[t];
    unsigned long long lt = (1ull << lane) - 1ull;
    int rank = 0;
#pragma unroll
    for (int ee = 0; ee < E_; ee++) {
        unsigned long long bm = __ballot(e == ee);
        if (e == ee) rank = __popcll(bm & lt);
    }
    int loc = chunkBase[c * E_ + e] + rank;
    bool keep = loc < CAP_;
    gate_val[t] = keep ? prob[t] : 0.f;
    int slot = e * CAP_ + loc;
    slot_keep[t] = keep ? slot : -1;
    if (keep) slot2tok[slot] = t;
}

__global__ __launch_bounds__(256) void k_dispatch(const unsigned short* __restrict__ xn,
                                                  const int* __restrict__ slot_keep,
                                                  unsigned short* __restrict__ disp) {
    int t = blockIdx.x;
    int s = slot_keep[t];
    if (s < 0) return;
    const uint2* src = (const uint2*)(xn + (size_t)t * D_);
    uint2* dst = (uint2*)(disp + (size_t)s * D_);
    dst[threadIdx.x] = src[threadIdx.x];   // 256 threads x 8B = 1024 bf16
}

__global__ void k_copy_f32(const float* __restrict__ src, float* __restrict__ dst, size_t n) {
    size_t i = (size_t)blockIdx.x * blockDim.x + threadIdx.x;
    size_t stride = (size_t)gridDim.x * blockDim.x;
    for (; i < n; i += stride) dst[i] = src[i];
}

__global__ void k_laux(const float* __restrict__ sums, float* __restrict__ out) {
    if (threadIdx.x == 0 && blockIdx.x == 0) {
        float s = 0.f;
        for (int e = 0; e < E_; e++) s += sums[e] * sums[8 + e];
        out[0] = (float)E_ * s / ((float)N_ * (float)N_);
    }
}

// ---------------- host orchestration ----------------
extern "C" void kernel_launch(void* const* d_in, const int* in_sizes, int n_in,
                              void* d_out, int out_size, void* d_ws, size_t ws_size,
                              hipStream_t stream) {
    (void)in_sizes; (void)n_in; (void)out_size; (void)ws_size;
    const float* x     = (const float*)d_in[0];
    // d_in[1] = attn_mask (causal, applied analytically)
    const float* wq    = (const float*)d_in[2];
    const float* bq    = (const float*)d_in[3];
    const float* wk    = (const float*)d_in[4];
    const float* bk    = (const float*)d_in[5];
    const float* wv    = (const float*)d_in[6];
    const float* bv    = (const float*)d_in[7];
    const float* wo    = (const float*)d_in[8];
    const float* bo    = (const float*)d_in[9];
    const float* ln1g  = (const float*)d_in[10];
    const float* ln1b  = (const float*)d_in[11];
    const float* ln2g  = (const float*)d_in[12];
    const float* ln2b  = (const float*)d_in[13];
    const float* gatew = (const float*)d_in[14];
    const float* w1    = (const float*)d_in[15];
    const float* b1    = (const float*)d_in[16];
    const float* w2    = (const float*)d_in[17];
    const float* b2    = (const float*)d_in[18];

    char* p = (char*)d_ws;
    auto alloc = [&](size_t bytes) { void* r = (void*)p; p += (bytes + 255) & ~(size_t)255; return r; };
    auto* wq_bf = (unsigned short*)alloc((size_t)D_ * D_ * 2);
    auto* wk_bf = (unsigned short*)alloc((size_t)D_ * D_ * 2);
    auto* wv_bf = (unsigned short*)alloc((size_t)D_ * D_ * 2);
    auto* wo_bf = (unsigned short*)alloc((size_t)D_ * D_ * 2);
    auto* gw_bf = (unsigned short*)alloc((size_t)D_ * E_ * 2);
    auto* w1_bf = (unsigned short*)alloc((size_t)E_ * D_ * F_ * 2);
    auto* w2_bf = (unsigned short*)alloc((size_t)E_ * F_ * D_ * 2);
    auto* xn_bf = (unsigned short*)alloc((size_t)N_ * D_ * 2);
    auto* q_bf  = (unsigned short*)alloc((size_t)N_ * D_ * 2);
    auto* k_bf  = (unsigned short*)alloc((size_t)N_ * D_ * 2);
    auto* v_bf  = (unsigned short*)alloc((size_t)N_ * D_ * 2);
    auto* o_bf  = (unsigned short*)alloc((size_t)N_ * D_ * 2);
    auto* x1    = (float*)alloc((size_t)N_ * D_ * 4);
    auto* logits = (float*)alloc((size_t)N_ * E_ * 4);
    auto* eidx   = (int*)alloc((size_t)N_ * 4);
    auto* prob   = (float*)alloc((size_t)N_ * 4);
    auto* gate_val = (float*)alloc((size_t)N_ * 4);
    auto* slot_keep = (int*)alloc((size_t)N_ * 4);
    auto* waveCounts = (int*)alloc((size_t)(N_ / 32) * E_ * 4);
    auto* chunkBase  = (int*)alloc((size_t)(N_ / 32) * E_ * 4);
    auto* slot2tok   = (int*)alloc((size_t)E_ * CAP_ * 4);
    auto* sums       = (float*)alloc(64);
    auto* disp = (unsigned short*)alloc((size_t)E_ * CAP_ * D_ * 2);
    auto* h_bf = (unsigned short*)alloc((size_t)CAP_ * F_ * 2);

    auto cvt = [&](const float* s, unsigned short* d, size_t n) {
        k_f2bf<<<dim3(2048), dim3(256), 0, stream>>>(s, d, n);
    };
    cvt(wq, wq_bf, (size_t)D_ * D_);
    cvt(wk, wk_bf, (size_t)D_ * D_);
    cvt(wv, wv_bf, (size_t)D_ * D_);
    cvt(wo, wo_bf, (size_t)D_ * D_);
    cvt(gatew, gw_bf, (size_t)D_ * E_);
    cvt(w1, w1_bf, (size_t)E_ * D_ * F_);
    cvt(w2, w2_bf, (size_t)E_ * F_ * D_);

    k_init<<<dim3((E_ * CAP_ + 255) / 256), dim3(256), 0, stream>>>(slot2tok, sums);

    // ---- attention block ----
    k_ln<<<dim3(N_), dim3(256), 0, stream>>>(x, ln1g, ln1b, xn_bf);
    dim3 gproj(D_ / 128, N_ / 128);
    k_gemm<0><<<gproj, 256, 0, stream>>>(xn_bf, D_, wq_bf, D_, bq, 0.125f,
                                         nullptr, nullptr, q_bf, D_, nullptr, nullptr, D_);
    k_gemm<0><<<gproj, 256, 0, stream>>>(xn_bf, D_, wk_bf, D_, bk, 1.0f,
                                         nullptr, nullptr, k_bf, D_, nullptr, nullptr, D_);
    k_gemm<0><<<gproj, 256, 0, stream>>>(xn_bf, D_, wv_bf, D_, bv, 1.0f,
                                         nullptr, nullptr, v_bf, D_, nullptr, nullptr, D_);
    k_attn<<<dim3(S_ / 16, B_ * H_), 32, 0, stream>>>(q_bf, k_bf, v_bf, o_bf);
    k_gemm<1><<<gproj, 256, 0, stream>>>(o_bf, D_, wo_bf, D_, bo, 1.0f,
                                         x, x1, nullptr, D_, nullptr, nullptr, D_);

    // ---- MoE block ----
    k_ln<<<dim3(N_), dim3(256), 0, stream>>>(x1, ln2g, ln2b, xn_bf);
    k_gate_logits<<<dim3(N_ / 256), 256, 0, stream>>>(xn_bf, gw_bf, logits);
    k_gate_soft<<<dim3(N_ / 256), 256, 0, stream>>>(logits, eidx, prob, sums);
    k_count<<<dim3(N_ / 32), 32, 0, stream>>>(eidx, waveCounts);
    k_scan<<<dim3(1), dim3(8), 0, stream>>>(waveCounts, chunkBase);
    k_meta<<<dim3(N_ / 32), 32, 0, stream>>>(eidx, prob, chunkBase, gate_val, slot_keep, slot2tok);
    k_dispatch<<<dim3(N_), 256, 0, stream>>>(xn_bf, slot_keep, disp);

    float* outF = (float*)d_out;
    k_copy_f32<<<dim3(4096), 256, 0, stream>>>(x1, outF, (size_t)N_ * D_);

    for (int e = 0; e < E_; e++) {
        k_gemm<2><<<dim3(F_ / 128, CAP_ / 128), 256, 0, stream>>>(
            disp + (size_t)e * CAP_ * D_, D_,
            w1_bf + (size_t)e * D_ * F_, F_,
            b1 + (size_t)e * F_, 1.0f,
            nullptr, nullptr, h_bf, F_, nullptr, nullptr, D_);
        k_gemm<3><<<dim3(D_ / 128, CAP_ / 128), 256, 0, stream>>>(
            h_bf, F_,
            w2_bf + (size_t)e * F_ * D_, D_,
            b2 + (size_t)e * D_, 1.0f,
            nullptr, outF, nullptr, D_,
            slot2tok + (size_t)e * CAP_, gate_val, F_);
    }

    k_laux<<<dim3(1), dim3(1), 0, stream>>>(sums, outF + (size_t)N_ * D_);
}
